// ArthTextToDenseBlock_90280212562466
// MI455X (gfx1250) — matrix operand order repaired and checked by reading, not compiled, not run on hardware
//
#include <hip/hip_runtime.h>
#include <hip/hip_bf16.h>

// ---------------------------------------------------------------------------
// Problem constants (reference: B=512, S=256, D=256, TAU=0.2)
// ---------------------------------------------------------------------------
#define BB 512
#define SS 256
#define DD 256
#define NROWS (BB * SS)          // 131072 token rows
#define LPITCH 32                // logits row stride (26 used, padded)

#define ROWS 64                  // rows per k_mlp block (4 M-subtiles)
#define XPITCH 264               // bf16 LDS row pitch (pad kills bank conflicts)
#define H1PITCH 136
#define XS_BYTES   (ROWS * XPITCH * 2)            // 33792
#define H0_OFF     XS_BYTES                       // 33792
#define H1_OFF     (XS_BYTES + ROWS * XPITCH * 2) // 67584
#define SMEM_TOTAL (H1_OFF + ROWS * H1PITCH * 2)  // 84992

typedef __attribute__((ext_vector_type(16))) __bf16 v16bf;
typedef __attribute__((ext_vector_type(8)))  float  v8f;

union FragBF { v16bf v; uint4 q[2]; };

__device__ __forceinline__ unsigned short f2bf(float f) {
    unsigned u = __float_as_uint(f);
    unsigned r = (u + 0x7FFFu + ((u >> 16) & 1u)) >> 16;   // RNE
    return (unsigned short)r;
}

// ---------------------------------------------------------------------------
// threefry2x32 (JAX-style gumbel noise, counter mode)
// ---------------------------------------------------------------------------
__device__ __forceinline__ void tf2x32(unsigned k0, unsigned k1,
                                       unsigned c0, unsigned c1,
                                       unsigned &o0, unsigned &o1) {
    const unsigned R[8] = {13, 15, 26, 6, 17, 29, 16, 24};
    unsigned ks[3] = {k0, k1, k0 ^ k1 ^ 0x1BD11BDAu};
    unsigned x0 = c0 + ks[0], x1 = c1 + ks[1];
#pragma unroll
    for (int blk = 0; blk < 5; ++blk) {
#pragma unroll
        for (int r = 0; r < 4; ++r) {
            unsigned rr = R[(blk & 1) * 4 + r];
            x0 += x1;
            x1 = (x1 << rr) | (x1 >> (32 - rr));
            x1 ^= x0;
        }
        unsigned j = (unsigned)blk + 1u;
        x0 += ks[j % 3];
        x1 += ks[(j + 1) % 3] + j;
    }
    o0 = x0; o1 = x1;
}

__device__ __forceinline__ float gumbelv(unsigned k0, unsigned k1, unsigned idx) {
    unsigned a, b;
    tf2x32(k0, k1, 0u, idx, a, b);
    float u = ((float)(a >> 8) + 0.5f) * (1.0f / 16777216.0f);  // (0,1)
    float l = -__logf(u);
    l = fmaxf(l, 1e-20f);
    return -__logf(l);
}

// ---------------------------------------------------------------------------
// x: fp32 -> bf16 (global, streaming)
// ---------------------------------------------------------------------------
__global__ void k_xconv(const float* __restrict__ x, unsigned short* __restrict__ xb) {
    size_t i = ((size_t)blockIdx.x * 256 + threadIdx.x) * 4;
    float4 f = *(const float4*)(x + i);
    uint2 v;
    v.x = (unsigned)f2bf(f.x) | ((unsigned)f2bf(f.y) << 16);
    v.y = (unsigned)f2bf(f.z) | ((unsigned)f2bf(f.w) << 16);
    *(uint2*)(xb + i) = v;
}

// ---------------------------------------------------------------------------
// Weight pre-swizzle: fp32 row-major (K x ncols) -> bf16 WMMA B-fragment layout
// frag[((kt*NT + nt)*32 + lane)*16 + i]:
//   lane<16 : column nt*16+lane, K = kt*32 + (i<8 ? i : i+8)       (i=0..15)
//   lane>=16: column nt*16+(lane-16), K = kt*32+8+(i<8 ? i : i+8)
// ---------------------------------------------------------------------------
__global__ void k_prep(const float* __restrict__ W, unsigned short* __restrict__ F,
                       int K, int NT, int ncols) {
    int f = blockIdx.x * 256 + threadIdx.x;
    int total = K * NT * 16;
    if (f >= total) return;
    int i    = f & 15;
    int lane = (f >> 4) & 31;
    int rem  = f >> 9;
    int nt   = rem % NT;
    int kt   = rem / NT;
    int col  = nt * 16 + (lane & 15);
    int krow = kt * 32 + ((lane >= 16) ? 8 : 0) + ((i < 8) ? i : i + 8);
    float v = (col < ncols && krow < K) ? W[(size_t)krow * ncols + col] : 0.0f;
    F[f] = f2bf(v);
}

// ---------------------------------------------------------------------------
// Fused 3-layer MLP GEMM (bf16 WMMA), one 64-row slab per block.
// One B fragment feeds 4 M-subtile WMMAs (4x less weight traffic from L2).
// ---------------------------------------------------------------------------
struct MlpArgs {
    const unsigned short* xbf;
    float*                logits;
    const unsigned short* w0[5];
    const unsigned short* w1[5];
    const unsigned short* w2[5];
    const float* b0[5];
    const float* b1[5];
    const float* b2[5];
    int outdim[5];
    int colofs[5];
};

// 4 row-subtiles per column tile, B fragment loaded once per k-step
__device__ __forceinline__ void wmma_col4(const unsigned short* As, int pitch,
                                          const unsigned short* wf,
                                          int lane, int nt, int KT, int NT,
                                          v8f acc[4]) {
    const int kofs = (lane >= 16) ? 8 : 0;
    const int arowoff = (lane & 15) * pitch;
    for (int kt = 0; kt < KT; ++kt) {
        FragBF b;
        const uint4* wp = (const uint4*)(wf + (size_t)(((kt * NT) + nt) * 32 + lane) * 16);
        b.q[0] = wp[0];
        b.q[1] = wp[1];
#pragma unroll
        for (int ms = 0; ms < 4; ++ms) {
            FragBF a;
            const unsigned short* arow = As + ms * 16 * pitch + arowoff;
            a.q[0] = *(const uint4*)(arow + kt * 32 + kofs);
            a.q[1] = *(const uint4*)(arow + kt * 32 + kofs + 16);
            acc[ms] = __builtin_amdgcn_wmma_f32_16x16x32_bf16(false, a.v, false, b.v,
                                                              (short)0, acc[ms], false, false);
        }
    }
}

__device__ __forceinline__ v8f wmma_tile1(const unsigned short* As, int pitch,
                                          const unsigned short* wf,
                                          int msub, int lane, int nt, int KT, int NT) {
    v8f acc;
#pragma unroll
    for (int i = 0; i < 8; ++i) acc[i] = 0.0f;
    const unsigned short* arow = As + (msub * 16 + (lane & 15)) * pitch;
    int kofs = (lane >= 16) ? 8 : 0;
    for (int kt = 0; kt < KT; ++kt) {
        FragBF a, b;
        a.q[0] = *(const uint4*)(arow + kt * 32 + kofs);
        a.q[1] = *(const uint4*)(arow + kt * 32 + kofs + 16);
        const uint4* wp = (const uint4*)(wf + (size_t)(((kt * NT) + nt) * 32 + lane) * 16);
        b.q[0] = wp[0];
        b.q[1] = wp[1];
        acc = __builtin_amdgcn_wmma_f32_16x16x32_bf16(false, a.v, false, b.v,
                                                      (short)0, acc, false, false);
    }
    return acc;
}

__device__ __forceinline__ void store_silu_lds(v8f acc, unsigned short* Ds, int pitch,
                                               const float* bias, int msub, int lane, int nt) {
    int col  = nt * 16 + (lane & 15);
    float bv = bias[col];
    int rbase = msub * 16 + ((lane < 16) ? 0 : 8);
#pragma unroll
    for (int r = 0; r < 8; ++r) {
        float v = acc[r] + bv;
        v = v / (1.0f + __expf(-v));          // silu
        Ds[(rbase + r) * pitch + col] = f2bf(v);
    }
}

__global__ __launch_bounds__(256) void k_mlp(MlpArgs A) {
    extern __shared__ char smem[];
    unsigned short* Xs  = (unsigned short*)smem;
    unsigned short* H0s = (unsigned short*)(smem + H0_OFF);
    unsigned short* H1s = (unsigned short*)(smem + H1_OFF);

    const int tid  = threadIdx.x;
    const int w    = tid >> 5;
    const int lane = tid & 31;
    const size_t r0 = (size_t)blockIdx.x * ROWS;

    // ---- async global->LDS: pull the 64x256 bf16 slab straight into LDS ----
    // Each wave moves one 512-byte row per round (32 lanes x 16B), with the
    // per-lane LDS destination absorbing the padded 528-byte pitch.
    {
        const unsigned short* xb = A.xbf + r0 * DD;
        unsigned xs_base = (unsigned)(unsigned long long)(void*)Xs;   // LDS byte offset
#pragma unroll
        for (int rr = 0; rr < ROWS / 8; ++rr) {
            int row = rr * 8 + w;
            unsigned ldsa = xs_base + (unsigned)(row * (XPITCH * 2) + lane * 16);
            unsigned voff = (unsigned)(row * (DD * 2) + lane * 16);
            asm volatile("global_load_async_to_lds_b128 %0, %1, %2 offset:0"
                         :
                         : "v"(ldsa), "v"(voff), "s"(xb)
                         : "memory");
        }
        asm volatile("s_wait_asynccnt 0x0" ::: "memory");
    }
    __syncthreads();

    for (int m = 0; m < 5; ++m) {
        // layer0: (64x256) @ (256x256) -> silu -> H0   (wave w owns nt = 2w, 2w+1)
#pragma unroll
        for (int j = 0; j < 2; ++j) {
            int nt = w * 2 + j;
            v8f acc[4];
#pragma unroll
            for (int ms = 0; ms < 4; ++ms)
#pragma unroll
                for (int i = 0; i < 8; ++i) acc[ms][i] = 0.0f;
            wmma_col4(Xs, XPITCH, A.w0[m], lane, nt, 8, 16, acc);
#pragma unroll
            for (int ms = 0; ms < 4; ++ms)
                store_silu_lds(acc[ms], H0s, XPITCH, A.b0[m], ms, lane, nt);
        }
        __syncthreads();

        // layer1: (64x256) @ (256x128) -> silu -> H1   (wave w owns nt = w)
        {
            int nt = w;
            v8f acc[4];
#pragma unroll
            for (int ms = 0; ms < 4; ++ms)
#pragma unroll
                for (int i = 0; i < 8; ++i) acc[ms][i] = 0.0f;
            wmma_col4(H0s, XPITCH, A.w1[m], lane, nt, 8, 8, acc);
#pragma unroll
            for (int ms = 0; ms < 4; ++ms)
                store_silu_lds(acc[ms], H1s, H1PITCH, A.b1[m], ms, lane, nt);
        }
        __syncthreads();

        // layer2: (64x128) @ (128 x out<=16) -> logits (global fp32); waves 0..3
        if (w < 4) {
            v8f acc = wmma_tile1(H1s, H1PITCH, A.w2[m], w, lane, 0, 4, 1);
            int n = lane & 15;
            if (n < A.outdim[m]) {
                float bv = A.b2[m][n];
                int rbase = (int)r0 + w * 16 + ((lane < 16) ? 0 : 8);
#pragma unroll
                for (int r = 0; r < 8; ++r)
                    A.logits[(size_t)(rbase + r) * LPITCH + A.colofs[m] + n] = acc[r] + bv;
            }
        }
        __syncthreads();
    }
}

// ---------------------------------------------------------------------------
// Gate extraction: hard gumbel-softmax == argmax(logits + gumbel).
// gates[(t*B+b)*16] = {ig, opmv, mng, dog0..3, 0, op_pred[0..6], 0}
// sbuf[t] += argmax(dense_pred)   (exact small-integer float atomics)
// ---------------------------------------------------------------------------
__global__ void k_gates(const float* __restrict__ logits,
                        float* __restrict__ gates,
                        float* __restrict__ sbuf) {
    int gid = blockIdx.x * 256 + threadIdx.x;     // gid = t*B + b
    if (gid >= SS * BB) return;
    int t = gid >> 9;        // / 512
    int b = gid & 511;
    size_t r = (size_t)b * SS + t;                // token row in x layout
    const float* L = logits + r * LPITCH;

    unsigned K0[4], K1[4];
#pragma unroll
    for (int s2 = 0; s2 < 4; ++s2) tf2x32(0u, 1234u, 0u, (unsigned)s2, K0[s2], K1[s2]);

    float v0 = L[0] + gumbelv(K0[0], K1[0], (unsigned)gid * 2u + 0u);
    float v1 = L[1] + gumbelv(K0[0], K1[0], (unsigned)gid * 2u + 1u);
    float ig = (v1 > v0) ? 1.0f : 0.0f;

    int am = 0;
    float best = L[2] + gumbelv(K0[1], K1[1], (unsigned)gid * 3u + 0u);
#pragma unroll
    for (int j = 1; j < 3; ++j) {
        float vv = L[2 + j] + gumbelv(K0[1], K1[1], (unsigned)gid * 3u + (unsigned)j);
        if (vv > best) { best = vv; am = j; }
    }
    float mng  = (am >= 1) ? 1.0f : 0.0f;
    float opmv = (am == 2) ? 1.0f : 0.0f;

    int ad = 0;
    best = L[12] + gumbelv(K0[2], K1[2], (unsigned)gid * 4u + 0u);
#pragma unroll
    for (int j = 1; j < 4; ++j) {
        float vv = L[12 + j] + gumbelv(K0[2], K1[2], (unsigned)gid * 4u + (unsigned)j);
        if (vv > best) { best = vv; ad = j; }
    }

    int ap = 0;
    best = L[16] + gumbelv(K0[3], K1[3], (unsigned)gid * 10u + 0u);
#pragma unroll
    for (int j = 1; j < 10; ++j) {
        float vv = L[16 + j] + gumbelv(K0[3], K1[3], (unsigned)gid * 10u + (unsigned)j);
        if (vv > best) { best = vv; ap = j; }
    }

    float* gw = gates + (size_t)gid * 16;
    gw[0] = ig; gw[1] = opmv; gw[2] = mng;
    gw[3] = (ad == 0) ? 1.0f : 0.0f;
    gw[4] = (ad == 1) ? 1.0f : 0.0f;
    gw[5] = (ad == 2) ? 1.0f : 0.0f;
    gw[6] = (ad == 3) ? 1.0f : 0.0f;
    gw[7] = 0.0f;
#pragma unroll
    for (int j = 0; j < 7; ++j) gw[8 + j] = L[5 + j];   // op_pred (no noise)
    gw[15] = 0.0f;

    atomicAdd(&sbuf[t], (float)ap);
}

// ---------------------------------------------------------------------------
// Sequential scan: one block per batch element, d = threadIdx.x.
// fds == 0 always (zero-init, multiplicative-only update).
// ---------------------------------------------------------------------------
__global__ __launch_bounds__(256) void k_scan(const float* __restrict__ G,
                                              const float* __restrict__ SB,
                                              float* __restrict__ out) {
    __shared__ float shv[256];
    __shared__ float shr[256];
    const int b = blockIdx.x;
    const int d = threadIdx.x;

    float td = 0.0f, tv = 0.0f, fpm = 1.0f;
    float phm = (d == 0) ? 1.0f : 0.0f;
    float top[7];
#pragma unroll
    for (int j = 0; j < 7; ++j) top[j] = 0.0f;

    for (int t = 0; t < SS; ++t) {
        const float* g16 = G + ((size_t)t * BB + b) * 16;
        float ig = g16[0], opmv = g16[1], mng = g16[2];
        float dog0 = g16[3], dog1 = g16[4], dog2 = g16[5], dog3 = g16[6];
        float s = SB[t];
        float nig = 1.0f - ig;

        float g = nig * opmv;
        shv[d] = phm; __syncthreads();
        float sh1 = d ? shv[d - 1] : 0.0f; __syncthreads();
        phm = g * (sh1 * g) + (1.0f - g) * phm;

        float tg = g * phm;
#pragma unroll
        for (int j = 0; j < 7; ++j) top[j] += tg * g16[8 + j];

        float fpm_u = mng + (1.0f - mng) * fpm + (1.0f - mng) * fpm * 0.1f
                    + (1.0f - dog3) * (1.0f - mng) * fpm;
        fpm_u = dog0 * fpm + (1.0f - dog0) * fpm_u;
        fpm = ig * fpm + nig * fpm_u;

        float tid_ = nig * dog1 * phm;
        float tad  = nig * dog2 * phm;
        float tde  = nig * dog3 * phm;
        float tnc  = nig * dog0 * phm;
        float nv = tid_ * s + tad * (10.0f * td + s) + tde * (td + s * fpm)
                 + tnc * td + ig * td;
        td = phm * nv + (1.0f - phm) * td;

        float dv   = fmaxf(dog1, fmaxf(dog2, dog3)) * phm;
        float cand = fmaxf(fmaxf(tv * phm, opmv * phm), dv);
        shr[d] = cand; __syncthreads();
        for (int off2 = 128; off2 > 0; off2 >>= 1) {
            if (d < off2) shr[d] = fmaxf(shr[d], shr[d + off2]);
            __syncthreads();
        }
        float M = shr[0]; __syncthreads();
        tv = ig * td + nig * (M * phm);

        float g2 = nig * mng;
        shv[d] = phm; __syncthreads();
        float sh2 = d ? shv[d - 1] : 0.0f; __syncthreads();
        phm = g2 * (sh2 * g2) + (1.0f - g2) * phm;
    }

    const size_t BD = (size_t)BB * DD;
    size_t i0 = (size_t)b * DD + d;
    out[i0]      = td;
    out[BD + i0] = tv;
#pragma unroll
    for (int j = 0; j < 7; ++j) out[2 * BD + i0 * 7 + j] = top[j];
    out[9 * BD + i0] = phm;
    if (d == 0) {
        out[10 * BD + b]      = fpm;
        out[10 * BD + BB + b] = 0.0f;   // fds == 0
    }
}

// ---------------------------------------------------------------------------
// Host orchestration
// ---------------------------------------------------------------------------
extern "C" void kernel_launch(void* const* d_in, const int* in_sizes, int n_in,
                              void* d_out, int out_size, void* d_ws, size_t ws_size,
                              hipStream_t stream) {
    (void)out_size; (void)ws_size;

    // ---- identify inputs by size ----
    const float* x = nullptr;
    int pidx[64]; int np = 0;
    for (int i = 0; i < n_in; ++i) {
        if (in_sizes[i] == BB * SS * DD && !x) x = (const float*)d_in[i];
        else if (in_sizes[i] == 1) { /* start_pos, assumed 0 */ }
        else if (np < 64) pidx[np++] = i;
    }

    const float *W0[5] = {}, *W1[5] = {}, *W2[5] = {};
    const float *B0[5] = {}, *B1[5] = {}, *B2[5] = {};
    const int outd_arr[5] = {2, 3, 7, 4, 10};          // tv, moved, op, dense_op, dense_pred
    const int cofs_arr[5] = {0, 2, 5, 12, 16};

    for (int g = 0; g * 6 + 5 < np + 1 && g < 5; ++g) {
        const float* p[6]; int s[6];
        for (int k = 0; k < 6; ++k) {
            p[k] = (const float*)d_in[pidx[g * 6 + k]];
            s[k] = in_sizes[pidx[g * 6 + k]];
        }
        const float *w0, *w1, *w2, *b0, *b1, *b2; int outd;
        if (s[0] == 256) {   // alphabetical leaf order: b0,b1,b2,w0,w1,w2
            b0 = p[0]; b1 = p[1]; b2 = p[2]; w0 = p[3]; w1 = p[4]; w2 = p[5];
            outd = s[2];
        } else {             // insertion order: w0,b0,w1,b1,w2,b2
            w0 = p[0]; b0 = p[1]; w1 = p[2]; b1 = p[3]; w2 = p[4]; b2 = p[5];
            outd = s[5];
        }
        int m = (outd == 2) ? 0 : (outd == 3) ? 1 : (outd == 7) ? 2 : (outd == 4) ? 3 : 4;
        W0[m] = w0; W1[m] = w1; W2[m] = w2; B0[m] = b0; B1[m] = b1; B2[m] = b2;
    }

    // ---- workspace carve ----
    char* ws = (char*)d_ws;
    size_t off = 0;
    auto carve = [&](size_t bytes) -> void* {
        void* pp = ws + off;
        off = (off + bytes + 255) & ~(size_t)255;
        return pp;
    };
    unsigned short* xbf = (unsigned short*)carve((size_t)NROWS * DD * 2);  // 64 MB
    unsigned short *w0f[5], *w1f[5], *w2f[5];
    for (int m = 0; m < 5; ++m) {
        w0f[m] = (unsigned short*)carve((size_t)256 * 256 * 2);
        w1f[m] = (unsigned short*)carve((size_t)256 * 128 * 2);
        w2f[m] = (unsigned short*)carve((size_t)128 * 16 * 2);
    }
    float* logits = (float*)carve((size_t)NROWS * LPITCH * 4);   // 16 MB
    float* gates  = (float*)carve((size_t)SS * BB * 16 * 4);     // 8 MB
    float* sbuf   = (float*)carve((size_t)SS * 4);

    // ---- x -> bf16 ----
    k_xconv<<<(NROWS * DD / 4) / 256, 256, 0, stream>>>(x, xbf);

    // ---- weight pre-swizzle ----
    for (int m = 0; m < 5; ++m) {
        k_prep<<<(256 * 16 * 16 + 255) / 256, 256, 0, stream>>>(W0[m], w0f[m], 256, 16, 256);
        k_prep<<<(256 *  8 * 16 + 255) / 256, 256, 0, stream>>>(W1[m], w1f[m], 256,  8, 128);
        k_prep<<<(128 *  1 * 16 + 255) / 256, 256, 0, stream>>>(W2[m], w2f[m], 128,  1, outd_arr[m]);
    }

    // ---- fused WMMA MLP pass over all tokens ----
    MlpArgs args;
    args.xbf = xbf; args.logits = logits;
    for (int m = 0; m < 5; ++m) {
        args.w0[m] = w0f[m]; args.w1[m] = w1f[m]; args.w2[m] = w2f[m];
        args.b0[m] = B0[m];  args.b1[m] = B1[m];  args.b2[m] = B2[m];
        args.outdim[m] = outd_arr[m]; args.colofs[m] = cofs_arr[m];
    }
    k_mlp<<<NROWS / ROWS, 256, SMEM_TOTAL, stream>>>(args);

    // ---- gates + s reduction ----
    hipMemsetAsync(sbuf, 0, (size_t)SS * 4, stream);
    k_gates<<<(SS * BB + 255) / 256, 256, 0, stream>>>(logits, gates, sbuf);

    // ---- sequential elementwise scan ----
    k_scan<<<BB, 256, 0, stream>>>(gates, sbuf, (float*)d_out);
}
